// MQ_25443386262158
// MI455X (gfx1250) — matrix operand order repaired and checked by reading, not compiled, not run on hardware
//
#include <hip/hip_runtime.h>

#define N_ 16384
#define E_ 512
#define DIM_ 128
#define K_ 4096
#define M_ 4
#define EPS_ 1e-5f

typedef __attribute__((ext_vector_type(16))) __bf16 v16bf;
typedef __attribute__((ext_vector_type(8)))  __bf16 bf16x8;
typedef __attribute__((ext_vector_type(8)))  float  v8f;
typedef __attribute__((ext_vector_type(4)))  float  f32x4;
typedef __attribute__((ext_vector_type(4)))  int    i32x4;

union FragAB { v16bf v; f32x4 f4[2]; };

typedef __attribute__((address_space(1))) i32x4 g_i32x4;   // global int4
typedef __attribute__((address_space(3))) i32x4 l_i32x4;   // LDS int4

#ifndef __has_builtin
#define __has_builtin(x) 0
#endif

// 16-byte global -> LDS copy: async on gfx1250 when available, else direct.
__device__ __forceinline__ void async_cp16(const void* g, void* l)
{
#if defined(__AMDGCN__) && __has_builtin(__builtin_amdgcn_global_load_async_to_lds_b128)
    __builtin_amdgcn_global_load_async_to_lds_b128(
        (g_i32x4*)g, (l_i32x4*)l, 0, 0);
#else
    *(f32x4*)l = *(const f32x4*)g;
#endif
}
__device__ __forceinline__ void async_wait()
{
#if defined(__AMDGCN__) && __has_builtin(__builtin_amdgcn_global_load_async_to_lds_b128)
    asm volatile("s_wait_asynccnt 0x0" ::: "memory");
#endif
}

// ---------------------------------------------------------------------------
// Build head-m input x_m (bf16) from the closed form of the in-place updates.
// ---------------------------------------------------------------------------
__global__ void mq_prep_x(const float* __restrict__ x, const int* __restrict__ masks,
                          const float* __restrict__ pos, __bf16* __restrict__ xb, int m)
{
    int i = blockIdx.x * blockDim.x + threadIdx.x;   // over N*E
    int row = i >> 9;
    int j   = i & 511;
    float p = pos[j];
    int mw = j >> 7;
    bool bound = ((j & 127) == 127);
    float v;
    if (row == 0) {
        bool mb = masks[(size_t)m * E_ + j] != 0;
        bool inwin = mb && (mw == m) && !bound;
        v = ((mb && !inwin) ? 1.0f : 0.0f) + p;
    } else {
        bool mb = masks[(size_t)mw * E_ + j] != 0;
        bool zeroed = mb && !bound && (m >= mw);
        v = zeroed ? (float)(m - mw + 1) * p
                   : x[(size_t)row * E_ + j] + (float)(m + 1) * p;
    }
    xb[i] = (__bf16)v;
}

// ---------------------------------------------------------------------------
// Coalesced column stats, stage 1: per-block partial sum/sumsq.
// ---------------------------------------------------------------------------
__global__ __launch_bounds__(256)
void mq_colpart(const float* __restrict__ Y, int C,
                float* __restrict__ psum, float* __restrict__ psq)
{
    __shared__ float s1[256], s2[256];
    const int t = threadIdx.x;
    const int c = t % C;         // C is 128 or 256
    const int rl = t / C;
    const int RL = 256 / C;
    const int rowBase = blockIdx.x * (N_ / 8);
    float sum = 0.f, sq = 0.f;
    for (int i = rl; i < N_ / 8; i += RL) {
        float v = Y[(size_t)(rowBase + i) * C + c];
        sum += v; sq += v * v;
    }
    s1[t] = sum; s2[t] = sq;
    __syncthreads();
    if (rl == 0) {
        if (RL == 2) { sum += s1[t + C]; sq += s2[t + C]; }
        psum[blockIdx.x * C + c] = sum;
        psq[blockIdx.x * C + c]  = sq;
    }
}

// stage 2: finalize BN scale/shift:  BN(y) = y*scale + shift
__global__ void mq_colfin(const float* __restrict__ psum, const float* __restrict__ psq,
                          int C, const float* __restrict__ g, const float* __restrict__ be,
                          float* __restrict__ scale, float* __restrict__ shift)
{
    int c = threadIdx.x;
    if (c >= C) return;
    float s = 0.f, q = 0.f;
    #pragma unroll
    for (int b = 0; b < 8; ++b) { s += psum[b * C + c]; q += psq[b * C + c]; }
    float mu  = s / (float)N_;
    float var = q / (float)N_ - mu * mu;
    float sc  = g[c] * rsqrtf(var + EPS_);
    scale[c] = sc;
    shift[c] = be[c] - mu * sc;
}

// ---------------------------------------------------------------------------
// WMMA GEMM: Y[N,Cout] = f(A)[N,Kd] @ W[Cout,Kd]^T + bias
//   MODE 0: A is bf16 (async-staged)   MODE 1: A is f32
//   MODE 2: A is f32, apply relu(A*scale+shift)
// Tile 128x64, 8 waves; each wave: 16 rows x 64 cols = 4 WMMAs per k-step.
// ---------------------------------------------------------------------------
template <int MODE, bool WBF>
__global__ __launch_bounds__(256)
void mq_gemm(const void* __restrict__ Ap, int lda,
             const float* __restrict__ scale, const float* __restrict__ shift,
             const float* __restrict__ W, const float* __restrict__ bias,
             float* __restrict__ Y, __bf16* __restrict__ Ybf,
             int Kd, int Cout)
{
    __shared__ __align__(16) __bf16 As[128 * 32];  // 8 KB
    __shared__ __align__(16) __bf16 Bs[64 * 32];   // 4 KB, stored [col][k]
    const int t = threadIdx.x;
    const int wave = t >> 5, lane = t & 31;
    const int lr = lane & 15, hi = lane >> 4;
    const int rowBase = blockIdx.x * 128;
    const int colBase = blockIdx.y * 64;
    v8f acc[4] = {v8f{}, v8f{}, v8f{}, v8f{}};

    const int ra  = t >> 1;          // A staging row (0..127)
    const int ka  = (t & 1) * 16;    // A staging k   (0,16)
    const int rb  = t >> 2;          // B staging col (0..63)
    const int kb_ = (t & 3) * 8;     // B staging k   (0,8,16,24)

    for (int k0 = 0; k0 < Kd; k0 += 32) {
        // ---- stage A (128x32 bf16), 16 elems/thread ----
        if constexpr (MODE == 0) {
            const __bf16* A = (const __bf16*)Ap;
            const __bf16* g = A + (size_t)(rowBase + ra) * lda + k0 + ka;
            async_cp16(g,     As + ra * 32 + ka);
            async_cp16(g + 8, As + ra * 32 + ka + 8);
        } else {
            const float* A = (const float*)Ap;
            const f32x4* src = (const f32x4*)(A + (size_t)(rowBase + ra) * lda + k0 + ka);
            f32x4 u0 = src[0], u1 = src[1], u2 = src[2], u3 = src[3];
            if constexpr (MODE == 2) {
                int c = k0 + ka;
                f32x4 s0 = *(const f32x4*)(scale + c),     s1 = *(const f32x4*)(scale + c + 4);
                f32x4 s2 = *(const f32x4*)(scale + c + 8), s3 = *(const f32x4*)(scale + c + 12);
                f32x4 h0 = *(const f32x4*)(shift + c),     h1 = *(const f32x4*)(shift + c + 4);
                f32x4 h2 = *(const f32x4*)(shift + c + 8), h3 = *(const f32x4*)(shift + c + 12);
                u0 = u0 * s0 + h0; u1 = u1 * s1 + h1;
                u2 = u2 * s2 + h2; u3 = u3 * s3 + h3;
                #pragma unroll
                for (int j = 0; j < 4; ++j) {
                    u0[j] = fmaxf(u0[j], 0.f); u1[j] = fmaxf(u1[j], 0.f);
                    u2[j] = fmaxf(u2[j], 0.f); u3[j] = fmaxf(u3[j], 0.f);
                }
            }
            bf16x8 o0, o1;
            #pragma unroll
            for (int j = 0; j < 4; ++j) {
                o0[j] = (__bf16)u0[j]; o0[4 + j] = (__bf16)u1[j];
                o1[j] = (__bf16)u2[j]; o1[4 + j] = (__bf16)u3[j];
            }
            *(bf16x8*)(As + ra * 32 + ka)     = o0;
            *(bf16x8*)(As + ra * 32 + ka + 8) = o1;
        }
        // ---- stage B (64 cols x 32 k, from f32 W), 8 elems/thread ----
        {
            const f32x4* wsrc = (const f32x4*)(W + (size_t)(colBase + rb) * Kd + k0 + kb_);
            f32x4 u0 = wsrc[0], u1 = wsrc[1];
            bf16x8 o;
            #pragma unroll
            for (int j = 0; j < 4; ++j) { o[j] = (__bf16)u0[j]; o[4 + j] = (__bf16)u1[j]; }
            *(bf16x8*)(Bs + rb * 32 + kb_) = o;
        }
        if constexpr (MODE == 0) async_wait();
        __syncthreads();

        FragAB a, b[4];
        {
            int ar = wave * 16 + lr;
            int kh = hi ? 8 : 0;
            a.f4[0] = *(const f32x4*)(As + ar * 32 + kh);
            a.f4[1] = *(const f32x4*)(As + ar * 32 + kh + 16);
            int kb = hi ? 8 : 0;
            #pragma unroll
            for (int s = 0; s < 4; ++s) {
                int c = s * 16 + lr;
                b[s].f4[0] = *(const f32x4*)(Bs + c * 32 + kb);
                b[s].f4[1] = *(const f32x4*)(Bs + c * 32 + kb + 16);
            }
        }
        #pragma unroll
        for (int s = 0; s < 4; ++s)
            acc[s] = __builtin_amdgcn_wmma_f32_16x16x32_bf16(false, a.v, false, b[s].v,
                                                             (short)0, acc[s], false, false);
        __syncthreads();
    }

    #pragma unroll
    for (int s = 0; s < 4; ++s) {
        int col = colBase + s * 16 + lr;
        float bb = bias[col];
        #pragma unroll
        for (int rr = 0; rr < 8; ++rr) {
            int row = rowBase + wave * 16 + hi * 8 + rr;
            float v = acc[s][rr] + bb;
            Y[(size_t)row * Cout + col] = v;
            if constexpr (WBF) Ybf[(size_t)row * Cout + col] = (__bf16)v;
        }
    }
}

// ---------------------------------------------------------------------------
// Codebook prep: bf16 copy + ||E_k||^2
// ---------------------------------------------------------------------------
__global__ __launch_bounds__(128)
void mq_prep_cb(const float* __restrict__ cb, __bf16* __restrict__ Eb,
                float* __restrict__ Esq, int m)
{
    __shared__ float s[128];
    int k = blockIdx.x;
    float v = cb[((size_t)m * K_ + k) * DIM_ + threadIdx.x];
    Eb[(size_t)k * DIM_ + threadIdx.x] = (__bf16)v;
    s[threadIdx.x] = v * v;
    __syncthreads();
    for (int off = 64; off > 0; off >>= 1) {
        if (threadIdx.x < off) s[threadIdx.x] += s[threadIdx.x + off];
        __syncthreads();
    }
    if (threadIdx.x == 0) Esq[k] = s[0];
}

// ---------------------------------------------------------------------------
// Fused distance GEMM + argmin over K=4096 codes; A (K=128) resident in regs,
// codebook double-buffered through LDS via async copies overlapping WMMAs.
// ---------------------------------------------------------------------------
__global__ __launch_bounds__(256)
void mq_dist(const __bf16* __restrict__ zeb, const __bf16* __restrict__ Eb,
             const float* __restrict__ Esq, int* __restrict__ idxOut)
{
    __shared__ __align__(16) __bf16 smem[128 * 128];   // 32 KB, reused
    const int t = threadIdx.x, wave = t >> 5, lane = t & 31;
    const int lr = lane & 15, hi = lane >> 4;
    const int rowBase = blockIdx.x * 128;

    {   // stage ze rows (128 x 128 bf16): 64 bf16 per thread, async
        int r = t >> 1, h = (t & 1) * 64;
        const __bf16* g = zeb + (size_t)(rowBase + r) * DIM_ + h;
        __bf16* l = smem + r * 128 + h;
        #pragma unroll
        for (int j = 0; j < 8; ++j) async_cp16(g + j * 8, l + j * 8);
    }
    async_wait();
    __syncthreads();

    FragAB a[4];
    {
        int r = wave * 16 + lr;
        int khb = hi ? 8 : 0;
        #pragma unroll
        for (int kc = 0; kc < 4; ++kc) {
            a[kc].f4[0] = *(const f32x4*)(smem + r * 128 + khb + 32 * kc);
            a[kc].f4[1] = *(const f32x4*)(smem + r * 128 + khb + 32 * kc + 16);
        }
    }
    __syncthreads();

    float minv[8]; int mini[8];
    #pragma unroll
    for (int r = 0; r < 8; ++r) { minv[r] = 3.4e38f; mini[r] = 0; }

    // double-buffered codebook chunks: 2 x (64 codes x 128 k) = 2 x 16 KB
    __bf16* buf0 = smem;
    __bf16* buf1 = smem + 64 * 128;
    const int cS = t >> 2, hS = (t & 3) * 32;   // staging: 32 bf16/thread

    auto stage = [&](int c0, __bf16* B) {
        const __bf16* g = Eb + (size_t)(c0 + cS) * DIM_ + hS;
        __bf16* l = B + cS * 128 + hS;
        #pragma unroll
        for (int j = 0; j < 4; ++j) async_cp16(g + j * 8, l + j * 8);
    };

    stage(0, buf0);
    async_wait();
    __syncthreads();

    for (int i = 0; i < K_ / 64; ++i) {
        int c0 = i * 64;
        if (i + 1 < K_ / 64) stage(c0 + 64, (i & 1) ? buf0 : buf1);  // prefetch next
        __bf16* Bs = (i & 1) ? buf1 : buf0;
        #pragma unroll
        for (int s = 0; s < 4; ++s) {
            FragAB b[4];
            int c  = 16 * s + lr;
            int kb = hi ? 8 : 0;
            #pragma unroll
            for (int kc = 0; kc < 4; ++kc) {
                b[kc].f4[0] = *(const f32x4*)(Bs + c * 128 + 32 * kc + kb);
                b[kc].f4[1] = *(const f32x4*)(Bs + c * 128 + 32 * kc + kb + 16);
            }
            v8f acc = {};
            #pragma unroll
            for (int kc = 0; kc < 4; ++kc)
                acc = __builtin_amdgcn_wmma_f32_16x16x32_bf16(false, a[kc].v, false, b[kc].v,
                                                              (short)0, acc, false, false);
            int col = c0 + 16 * s + lr;
            float e = Esq[col];
            #pragma unroll
            for (int r = 0; r < 8; ++r) {
                float d = e - 2.0f * acc[r];
                if (d < minv[r]) { minv[r] = d; mini[r] = col; }
            }
        }
        async_wait();       // next chunk's transfer has landed
        __syncthreads();    // everyone done reading Bs before it is overwritten
    }

    // cross-lane reduction: lanes 0..15 hold rows r, lanes 16..31 rows 8+r
    float* vbuf = (float*)smem;             // [8][32][8] floats = 8 KB
    int*   ibuf = (int*)(smem + 8192);      // 8 KB at byte offset 16 KB
    {
        float* vw = vbuf + (wave * 32 + lane) * 8;
        int*   iw = ibuf + (wave * 32 + lane) * 8;
        #pragma unroll
        for (int r = 0; r < 8; ++r) { vw[r] = minv[r]; iw[r] = mini[r]; }
    }
    __syncthreads();
    if (lane < 16) {
        int rr = lane;
        int h = rr >> 3, r = rr & 7;
        float bestv = 3.4e38f; int besti = 0x7fffffff;
        #pragma unroll
        for (int l = 0; l < 16; ++l) {
            int src = h * 16 + l;
            float v = vbuf[(wave * 32 + src) * 8 + r];
            int  ii = ibuf[(wave * 32 + src) * 8 + r];
            if (v < bestv || (v == bestv && ii < besti)) { bestv = v; besti = ii; }
        }
        idxOut[rowBase + wave * 16 + rr] = besti;
    }
}

// ---------------------------------------------------------------------------
// Gather ce = E[idx] (f32) into output, accumulate zq
// ---------------------------------------------------------------------------
__global__ __launch_bounds__(128)
void mq_gather(const int* __restrict__ idx, const float* __restrict__ cb,
               float* __restrict__ ceOut, float* __restrict__ zq, int m)
{
    int n = blockIdx.x, d = threadIdx.x;
    int k = idx[n];
    float v = cb[((size_t)m * K_ + k) * DIM_ + d];
    ceOut[(size_t)n * DIM_ + d] = v;
    zq[(size_t)n * DIM_ + d] += v;
}

// ---------------------------------------------------------------------------
extern "C" void kernel_launch(void* const* d_in, const int* in_sizes, int n_in,
                              void* d_out, int out_size, void* d_ws, size_t ws_size,
                              hipStream_t stream)
{
    const float* x    = (const float*)d_in[0];
    const int*   msk  = (const int*)  d_in[1];
    const float* pos  = (const float*)d_in[2];
    const float* cb   = (const float*)d_in[3];
    const float* ew1  = (const float*)d_in[4];
    const float* eb1  = (const float*)d_in[5];
    const float* eg1  = (const float*)d_in[6];
    const float* ebe1 = (const float*)d_in[7];
    const float* ew2  = (const float*)d_in[8];
    const float* eb2  = (const float*)d_in[9];
    const float* eg2  = (const float*)d_in[10];
    const float* ebe2 = (const float*)d_in[11];
    const float* ew3  = (const float*)d_in[12];
    const float* eb3  = (const float*)d_in[13];
    const float* dw1  = (const float*)d_in[14];
    const float* db1  = (const float*)d_in[15];
    const float* dg1  = (const float*)d_in[16];
    const float* dbe1 = (const float*)d_in[17];
    const float* dw2  = (const float*)d_in[18];
    const float* db2  = (const float*)d_in[19];
    const float* dg2  = (const float*)d_in[20];
    const float* dbe2 = (const float*)d_in[21];
    const float* dw3  = (const float*)d_in[22];
    const float* db3  = (const float*)d_in[23];

    float* xhat = (float*)d_out;                   // [N,E]
    float* res  = xhat + (size_t)N_ * E_;          // [M,N,DIM]
    float* ceo  = res + (size_t)M_ * N_ * DIM_;    // [M,N,DIM]

    char* w = (char*)d_ws;
    __bf16* xb   = (__bf16*)w;  w += (size_t)N_ * E_ * 2;
    float*  y1   = (float*)w;   w += (size_t)N_ * 128 * 4;
    float*  y2   = (float*)w;   w += (size_t)N_ * 256 * 4;
    __bf16* zeb  = (__bf16*)w;  w += (size_t)N_ * 128 * 2;
    __bf16* Ebuf = (__bf16*)w;  w += (size_t)K_ * DIM_ * 2;
    float*  Esq  = (float*)w;   w += (size_t)K_ * 4;
    int*    idx  = (int*)w;     w += (size_t)N_ * 4;
    float*  zq   = (float*)w;   w += (size_t)N_ * 128 * 4;
    float*  sc   = (float*)w;   w += 256 * 4;
    float*  sh   = (float*)w;   w += 256 * 4;
    float*  ps   = (float*)w;   w += 8 * 256 * 4;
    float*  pq   = (float*)w;   w += 8 * 256 * 4;

    (void)hipMemsetAsync(zq, 0, (size_t)N_ * 128 * 4, stream);

    for (int m = 0; m < M_; ++m) {
        mq_prep_x<<<(N_ * E_) / 256, 256, 0, stream>>>(x, msk, pos, xb, m);
        mq_gemm<0, false><<<dim3(N_ / 128, 128 / 64), 256, 0, stream>>>(
            xb, E_, nullptr, nullptr, ew1 + (size_t)m * 128 * E_, eb1 + m * 128,
            y1, nullptr, E_, 128);
        mq_colpart<<<8, 256, 0, stream>>>(y1, 128, ps, pq);
        mq_colfin<<<1, 256, 0, stream>>>(ps, pq, 128, eg1 + m * 128, ebe1 + m * 128, sc, sh);
        mq_gemm<2, false><<<dim3(N_ / 128, 256 / 64), 256, 0, stream>>>(
            y1, 128, sc, sh, ew2 + (size_t)m * 256 * 128, eb2 + m * 256,
            y2, nullptr, 128, 256);
        mq_colpart<<<8, 256, 0, stream>>>(y2, 256, ps, pq);
        mq_colfin<<<1, 256, 0, stream>>>(ps, pq, 256, eg2 + m * 256, ebe2 + m * 256, sc, sh);
        mq_gemm<2, true><<<dim3(N_ / 128, DIM_ / 64), 256, 0, stream>>>(
            y2, 256, sc, sh, ew3 + (size_t)m * DIM_ * 256, eb3 + m * DIM_,
            res + (size_t)m * N_ * DIM_, zeb, 256, DIM_);
        mq_prep_cb<<<K_, 128, 0, stream>>>(cb, Ebuf, Esq, m);
        mq_dist<<<N_ / 128, 256, 0, stream>>>(zeb, Ebuf, Esq, idx);
        mq_gather<<<N_, 128, 0, stream>>>(idx, cb, ceo + (size_t)m * N_ * DIM_, zq, m);
    }
    // decoder
    mq_gemm<1, false><<<dim3(N_ / 128, 256 / 64), 256, 0, stream>>>(
        zq, 128, nullptr, nullptr, dw1, db1, y2, nullptr, 128, 256);
    mq_colpart<<<8, 256, 0, stream>>>(y2, 256, ps, pq);
    mq_colfin<<<1, 256, 0, stream>>>(ps, pq, 256, dg1, dbe1, sc, sh);
    mq_gemm<2, false><<<dim3(N_ / 128, 128 / 64), 256, 0, stream>>>(
        y2, 256, sc, sh, dw2, db2, y1, nullptr, 256, 128);
    mq_colpart<<<8, 256, 0, stream>>>(y1, 128, ps, pq);
    mq_colfin<<<1, 256, 0, stream>>>(ps, pq, 128, dg2, dbe2, sc, sh);
    mq_gemm<2, false><<<dim3(N_ / 128, E_ / 64), 256, 0, stream>>>(
        y1, 128, sc, sh, dw3, db3, xhat, nullptr, 128, E_);
}
